// LSTM_65755949302210
// MI455X (gfx1250) — compile-verified, matrix operand order checked by reading, not run on hardware
//
#include <hip/hip_runtime.h>
#include <math.h>

// ---------------------------------------------------------------------------
// Problem constants
// ---------------------------------------------------------------------------
#define BB 32
#define TT 512
#define II 512
#define HH 1024
#define G4H 4096   // 4*H

typedef __bf16 bf16_t;
typedef __attribute__((ext_vector_type(16))) __bf16 v16bf;
typedef __attribute__((ext_vector_type(8)))  float  v8f;

union Frag32B {          // one WMMA A/B operand: 16 bf16 per lane = 32 bytes
    uint4 q[2];
    v16bf v;
};

__device__ __forceinline__ unsigned short f32_to_bf16(float f) {
    unsigned u = __float_as_uint(f);
    u += 0x7FFFu + ((u >> 16) & 1u);      // round-to-nearest-even
    return (unsigned short)(u >> 16);
}

__device__ __forceinline__ float sigmoidf_fast(float x) {
    return 1.0f / (1.0f + __expf(-x));
}

// ---------------------------------------------------------------------------
// f32 -> bf16 elementwise convert (weights)
// ---------------------------------------------------------------------------
__global__ __launch_bounds__(256)
void cvt_bf16_kernel(const float* __restrict__ in, unsigned short* __restrict__ out, int n) {
    int i = blockIdx.x * 256 + threadIdx.x;
    if (i < n) out[i] = f32_to_bf16(in[i]);
}

// ---------------------------------------------------------------------------
// x (B,T,I) f32  ->  xbf (T,B,I) bf16
// ---------------------------------------------------------------------------
__global__ __launch_bounds__(256)
void xpose_bf16_kernel(const float* __restrict__ x, unsigned short* __restrict__ o) {
    size_t i = (size_t)blockIdx.x * 256 + threadIdx.x;   // over T*B*I
    size_t t = i / ((size_t)BB * II);
    size_t r = i % ((size_t)BB * II);
    size_t b = r / II;
    size_t c = r % II;
    o[i] = f32_to_bf16(x[(b * TT + t) * II + c]);
}

// ---------------------------------------------------------------------------
// xg GEMM:  out[M=16384][4096] = A[M][K](bf16) * W[4096][K](bf16)^T + bih + bhh
// block tile 128(M) x 64(N), 256 threads = 8 waves, wave tile 32x32 (2x2 WMMA).
// K is a compile-time constant (512 / 1024) -> pure shift addressing.
// Software pipeline: tile k+1 fetched into registers while WMMAs run on tile k.
// ---------------------------------------------------------------------------
template <int K>
__global__ __launch_bounds__(256)
void gemm_xg_kernel(const unsigned short* __restrict__ A,
                    const unsigned short* __restrict__ W,
                    const float* __restrict__ bih,
                    const float* __restrict__ bhh,
                    float* __restrict__ out) {
    __shared__ unsigned short As[128 * 40];   // padded row stride 40 halves
    __shared__ unsigned short Bs[64 * 40];

    const int tid   = threadIdx.x;
    const int lane  = tid & 31;
    const int wv    = tid >> 5;
    const int mwave = wv & 3;        // 4 waves along M
    const int nwave = wv >> 2;       // 2 waves along N
    const int bm    = blockIdx.x;    // 128 blocks (M)
    const int bn    = blockIdx.y;    // 64 blocks (N)
    const int lrow  = lane & 15;
    const int sel   = (lane >> 4) << 3;   // 0 or 8 halves

    // fixed per-thread staging slots: 2 A-chunks (rows r, r+64) + 1 B-chunk
    const int srow = tid >> 2;                 // 0..63
    const int scol = (tid & 3) << 3;           // 0,8,16,24 halves
    const unsigned short* aG0 = A + (size_t)(bm * 128 + srow) * K + scol;
    const unsigned short* aG1 = A + (size_t)(bm * 128 + 64 + srow) * K + scol;
    const unsigned short* bG  = W + (size_t)(bn * 64 + srow) * K + scol;

    // prologue: fetch tile 0
    uint4 ra0 = *(const uint4*)aG0;
    uint4 ra1 = *(const uint4*)aG1;
    uint4 rb  = *(const uint4*)bG;

    v8f acc[2][2] = {};

    #pragma unroll 2
    for (int k0 = 0; k0 < K; k0 += 32) {
        __syncthreads();
        *(uint4*)(As + srow * 40 + scol)        = ra0;
        *(uint4*)(As + (64 + srow) * 40 + scol) = ra1;
        *(uint4*)(Bs + srow * 40 + scol)        = rb;
        __syncthreads();

        if (k0 + 32 < K) {    // issue next-tile fetch; latency hides under WMMAs
            ra0 = *(const uint4*)(aG0 + k0 + 32);
            ra1 = *(const uint4*)(aG1 + k0 + 32);
            rb  = *(const uint4*)(bG + k0 + 32);
        }

        Frag32B fa[2], fb[2];
        #pragma unroll
        for (int im = 0; im < 2; ++im) {
            const unsigned short* p = As + (mwave * 32 + im * 16 + lrow) * 40 + sel;
            fa[im].q[0] = *(const uint4*)(p);
            fa[im].q[1] = *(const uint4*)(p + 16);
        }
        #pragma unroll
        for (int jn = 0; jn < 2; ++jn) {
            const unsigned short* p = Bs + (nwave * 32 + jn * 16 + lrow) * 40 + sel;
            fb[jn].q[0] = *(const uint4*)(p);
            fb[jn].q[1] = *(const uint4*)(p + 16);
        }
        #pragma unroll
        for (int im = 0; im < 2; ++im)
            #pragma unroll
            for (int jn = 0; jn < 2; ++jn)
                acc[im][jn] = __builtin_amdgcn_wmma_f32_16x16x32_bf16(
                    false, fa[im].v, false, fb[jn].v, (short)0, acc[im][jn], false, false);
    }

    // epilogue: D layout — lane L: n = L%16, m = r + 8*(L>>4)
    const int hi = lane >> 4;
    #pragma unroll
    for (int im = 0; im < 2; ++im) {
        #pragma unroll
        for (int jn = 0; jn < 2; ++jn) {
            int n0 = bn * 64 + nwave * 32 + jn * 16 + lrow;
            int m0 = bm * 128 + mwave * 32 + im * 16 + 8 * hi;
            float bias = bih[n0] + bhh[n0];
            #pragma unroll
            for (int r = 0; r < 8; ++r)
                out[(size_t)(m0 + r) * G4H + n0] = acc[im][jn][r] + bias;
        }
    }
}

// ---------------------------------------------------------------------------
// grid-wide barrier (cooperative persistent kernel)
// ---------------------------------------------------------------------------
__device__ __forceinline__ void grid_sync(unsigned* cnt, unsigned target) {
    __threadfence();
    __syncthreads();
    if (threadIdx.x == 0) {
        __hip_atomic_fetch_add(cnt, 1u, __ATOMIC_ACQ_REL, __HIP_MEMORY_SCOPE_AGENT);
        while (__hip_atomic_load(cnt, __ATOMIC_ACQUIRE, __HIP_MEMORY_SCOPE_AGENT) < target) {
            __builtin_amdgcn_s_sleep(2);
        }
    }
    __syncthreads();
}

// ---------------------------------------------------------------------------
// Persistent LSTM recurrence. 64 WGs; WG wg owns H-columns [wg*16, wg*16+16).
// LDS (202 KB / WGP):
//   wslab : 4 gates x 16 rows x 1024 K bf16 (128 KB) resident all 512 steps
//   hs    : 32 x 1024 bf16 h broadcast slice (64 KB), restaged each step
//   gateT : 8 x 16 x 16 f32 gate tiles, cst: 32 x 16 f32 cell state
// 8 waves: wave = mt*4 + gate; each wave -> one 16x16 gate tile, K=1024 as
// 32 v_wmma_f32_16x16x32_bf16 in TWO independent accumulator chains
// (even/odd K chunks) fed purely by ds_load_b128.
// ---------------------------------------------------------------------------
__global__ __launch_bounds__(256)
void lstm_recurrent_kernel(const float* __restrict__ xg,            // (T*B, 4096) f32
                           const unsigned short* __restrict__ whh,  // (4096, 1024) bf16
                           unsigned short* __restrict__ hbuf,       // (2, 32, 1024) bf16
                           unsigned short* __restrict__ hseq,       // (T*B, 1024) bf16 or null
                           float* __restrict__ outseq,              // (B, T, H) f32 or null
                           float* __restrict__ hfin,                // (B, H) f32
                           float* __restrict__ cfin,                // (B, H) f32
                           unsigned* __restrict__ counter,
                           unsigned nblocks) {
    __shared__ unsigned short wslab[4 * 16 * 1024];   // 128 KB
    __shared__ unsigned short hs[32 * 1024];          // 64 KB
    __shared__ float gateT[8][16][16];                // 8 KB
    __shared__ float cst[32][16];                     // 2 KB

    const int tid  = threadIdx.x;
    const int lane = tid & 31;
    const int wv   = tid >> 5;
    const int wg   = blockIdx.x;          // 0..63
    const int lrow = lane & 15;
    const int sel  = (lane >> 4) << 3;    // 0 or 8 halves
    const int mt   = wv >> 2;             // batch half: rows mt*16..+16
    const int gate = wv & 3;              // i/f/g/o

    // preload w_hh slab: 64 rows x 1024 halves = 8192 16B chunks
    for (int c = tid; c < 64 * 128; c += 256) {
        int row = c >> 7, cc = (c & 127) << 3;
        int g = row >> 4, lr = row & 15;
        const uint4* src = (const uint4*)(whh + (size_t)(g * HH + wg * 16 + lr) * HH + cc);
        *(uint4*)(wslab + row * 1024 + cc) = *src;
    }
    for (int e = tid; e < 512; e += 256) cst[e >> 4][e & 15] = 0.0f;

    const unsigned short* bptr = wslab + (gate * 16 + lrow) * 1024 + sel;
    const unsigned short* aptr = hs + (mt * 16 + lrow) * 1024 + sel;

    for (int t = 0; t < TT; ++t) {
        // stage h slice into LDS: 64 KB = 4096 16B chunks, 16 per thread
        {
            const unsigned short* hsrc = hbuf + (size_t)(t & 1) * (BB * HH);
            #pragma unroll 4
            for (int c = tid; c < 4096; c += 256)
                *(uint4*)(hs + (c << 3)) = *(const uint4*)(hsrc + (c << 3));
        }
        // prefetch this step's xg gate rows (global_prefetch_b8) under the WMMAs
        {
            const float* xgr = xg + ((size_t)t * BB + (tid >> 4)) * G4H + wg * 16 + (tid & 15);
            __builtin_prefetch(xgr, 0, 1);
            __builtin_prefetch(xgr + HH, 0, 1);
            __builtin_prefetch(xgr + 2 * HH, 0, 1);
            __builtin_prefetch(xgr + 3 * HH, 0, 1);
            const float* xgr2 = xgr + 16ull * G4H;   // second batch half
            __builtin_prefetch(xgr2, 0, 1);
            __builtin_prefetch(xgr2 + HH, 0, 1);
            __builtin_prefetch(xgr2 + 2 * HH, 0, 1);
            __builtin_prefetch(xgr2 + 3 * HH, 0, 1);
        }
        __syncthreads();

        // two independent accumulator chains -> 2x WMMA ILP
        v8f acc0 = {};
        v8f acc1 = {};
        #pragma unroll 4
        for (int k0 = 0; k0 < HH; k0 += 64) {
            Frag32B fa0, fb0, fa1, fb1;
            fa0.q[0] = *(const uint4*)(aptr + k0);
            fa0.q[1] = *(const uint4*)(aptr + k0 + 16);
            fb0.q[0] = *(const uint4*)(bptr + k0);
            fb0.q[1] = *(const uint4*)(bptr + k0 + 16);
            fa1.q[0] = *(const uint4*)(aptr + k0 + 32);
            fa1.q[1] = *(const uint4*)(aptr + k0 + 48);
            fb1.q[0] = *(const uint4*)(bptr + k0 + 32);
            fb1.q[1] = *(const uint4*)(bptr + k0 + 48);
            acc0 = __builtin_amdgcn_wmma_f32_16x16x32_bf16(
                false, fa0.v, false, fb0.v, (short)0, acc0, false, false);
            acc1 = __builtin_amdgcn_wmma_f32_16x16x32_bf16(
                false, fa1.v, false, fb1.v, (short)0, acc1, false, false);
        }

        // scatter gate tile to LDS: lane L: n=L%16, m = r + 8*(L>>4)
        {
            const int hi = lane >> 4;
            #pragma unroll
            for (int r = 0; r < 8; ++r)
                gateT[wv][r + 8 * hi][lrow] = acc0[r] + acc1[r];
        }
        __syncthreads();

        // elementwise LSTM cell: 32 batch x 16 cols = 512 elems
        for (int e = tid; e < 512; e += 256) {
            int b = e >> 4, j = e & 15;
            int bmt = b >> 4, bl = b & 15;
            const float* xgr = xg + ((size_t)t * BB + b) * G4H + wg * 16 + j;
            float iv = gateT[bmt * 4 + 0][bl][j] + xgr[0];
            float fv = gateT[bmt * 4 + 1][bl][j] + xgr[HH];
            float gv = gateT[bmt * 4 + 2][bl][j] + xgr[2 * HH];
            float ov = gateT[bmt * 4 + 3][bl][j] + xgr[3 * HH];
            iv = sigmoidf_fast(iv);
            fv = sigmoidf_fast(fv);
            gv = tanhf(gv);
            ov = sigmoidf_fast(ov);
            float cnew = fv * cst[b][j] + iv * gv;
            float hnew = ov * tanhf(cnew);
            cst[b][j] = cnew;
            int col = wg * 16 + j;
            hbuf[(size_t)((t + 1) & 1) * (BB * HH) + (size_t)b * HH + col] = f32_to_bf16(hnew);
            if (hseq)   hseq[((size_t)t * BB + b) * HH + col] = f32_to_bf16(hnew);
            if (outseq) outseq[(size_t)b * TT * HH + (size_t)t * HH + col] = hnew;
            if (t == TT - 1) {
                hfin[(size_t)b * HH + col] = hnew;
                cfin[(size_t)b * HH + col] = cnew;
            }
        }

        grid_sync(counter, (unsigned)(t + 1) * nblocks);
    }
}

// ---------------------------------------------------------------------------
// host launcher
// ---------------------------------------------------------------------------
extern "C" void kernel_launch(void* const* d_in, const int* in_sizes, int n_in,
                              void* d_out, int out_size, void* d_ws, size_t ws_size,
                              hipStream_t stream) {
    (void)in_sizes; (void)n_in; (void)out_size; (void)ws_size;

    const float* x     = (const float*)d_in[0];
    const float* w_ih0 = (const float*)d_in[1];
    const float* w_hh0 = (const float*)d_in[2];
    const float* b_ih0 = (const float*)d_in[3];
    const float* b_hh0 = (const float*)d_in[4];
    const float* w_ih1 = (const float*)d_in[5];
    const float* w_hh1 = (const float*)d_in[6];
    const float* b_ih1 = (const float*)d_in[7];
    const float* b_hh1 = (const float*)d_in[8];
    float* out = (float*)d_out;

    // workspace carve-up (256B aligned)
    char* wsb = (char*)d_ws;
    size_t off = 0;
    auto take = [&](size_t bytes) -> void* {
        void* p = wsb + off;
        off += (bytes + 255) & ~(size_t)255;
        return p;
    };
    const size_t MROWS = (size_t)TT * BB;                       // 16384
    float*          xg    = (float*)take(MROWS * G4H * sizeof(float));          // 268 MB
    unsigned short* xbf   = (unsigned short*)take(MROWS * II * 2);              // 16 MB
    unsigned short* hseq  = (unsigned short*)take(MROWS * HH * 2);              // 33.5 MB
    unsigned short* wih0b = (unsigned short*)take((size_t)G4H * II * 2);
    unsigned short* whh0b = (unsigned short*)take((size_t)G4H * HH * 2);
    unsigned short* wih1b = (unsigned short*)take((size_t)G4H * HH * 2);
    unsigned short* whh1b = (unsigned short*)take((size_t)G4H * HH * 2);
    unsigned short* hbuf  = (unsigned short*)take((size_t)2 * BB * HH * 2);
    unsigned*       cnt   = (unsigned*)take(256);

    float* hfin0 = out + (size_t)BB * TT * HH;
    float* hfin1 = hfin0 + (size_t)BB * HH;
    float* cfin0 = hfin1 + (size_t)BB * HH;
    float* cfin1 = cfin0 + (size_t)BB * HH;

    // ---- weight / input conversions to bf16 ----
    {
        int n;
        n = G4H * II; cvt_bf16_kernel<<<(n + 255) / 256, 256, 0, stream>>>(w_ih0, wih0b, n);
        n = G4H * HH; cvt_bf16_kernel<<<(n + 255) / 256, 256, 0, stream>>>(w_hh0, whh0b, n);
        n = G4H * HH; cvt_bf16_kernel<<<(n + 255) / 256, 256, 0, stream>>>(w_ih1, wih1b, n);
        n = G4H * HH; cvt_bf16_kernel<<<(n + 255) / 256, 256, 0, stream>>>(w_hh1, whh1b, n);
    }
    {
        size_t n = MROWS * II;   // (B,T,I) -> (T,B,I) bf16
        xpose_bf16_kernel<<<(unsigned)(n / 256), 256, 0, stream>>>(x, xbf);
    }

    // ---- layer 0 ----
    hipMemsetAsync(hbuf, 0, (size_t)2 * BB * HH * 2, stream);
    hipMemsetAsync(cnt, 0, 256, stream);
    gemm_xg_kernel<II><<<dim3(128, 64), 256, 0, stream>>>(xbf, wih0b, b_ih0, b_hh0, xg);
    lstm_recurrent_kernel<<<64, 256, 0, stream>>>(xg, whh0b, hbuf, hseq, nullptr,
                                                  hfin0, cfin0, cnt, 64u);

    // ---- layer 1 ----
    gemm_xg_kernel<HH><<<dim3(128, 64), 256, 0, stream>>>(hseq, wih1b, b_ih1, b_hh1, xg);
    hipMemsetAsync(hbuf, 0, (size_t)2 * BB * HH * 2, stream);
    hipMemsetAsync(cnt, 0, 256, stream);
    lstm_recurrent_kernel<<<64, 256, 0, stream>>>(xg, whh1b, hbuf, nullptr, out,
                                                  hfin1, cfin1, cnt, 64u);
}
